// GCN_29583734735280
// MI455X (gfx1250) — matrix operand order repaired and checked by reading, not compiled, not run on hardware
//
#include <hip/hip_runtime.h>
#include <hip/hip_bf16.h>

// ---------------------------------------------------------------------------
// GCN (3-layer GraphConv) for MI455X / gfx1250.
//   - GEMMs (65.5 GFLOP) via V_WMMA_F32_16X16X32_BF16 (bf16 in, fp32 acc)
//     * 4-wave blocks, 128x64 block tile, 32x64 wave tile (8 WMMA / K-step)
//     * double-buffered LDS, register-staged global prefetch, 1 barrier/K-step
//   - edge gather/scatter via fp32 global atomics (h is 102MB -> L2-resident)
//   - degree norms, bias, ReLU in fp32
// ---------------------------------------------------------------------------

typedef __attribute__((ext_vector_type(16))) __bf16 v16bf;
typedef __attribute__((ext_vector_type(8)))  __bf16 v8bf;
typedef __attribute__((ext_vector_type(4)))  __bf16 v4bf;
typedef __attribute__((ext_vector_type(8)))  float  v8f;

union V16U { v16bf v; v8bf h[2]; };
static __device__ inline v16bf mk16(v8bf lo, v8bf hi) {
    V16U u; u.h[0] = lo; u.h[1] = hi; return u.v;
}

// ------------------------------ degree / norm ------------------------------

__global__ void degree_kernel(const int* __restrict__ src, const int* __restrict__ dst,
                              float* __restrict__ outdeg, float* __restrict__ indeg, int E) {
    int e = blockIdx.x * blockDim.x + threadIdx.x;
    if (e < E) {
        atomicAdd(&outdeg[src[e]], 1.0f);
        atomicAdd(&indeg[dst[e]], 1.0f);
    }
}

__global__ void norm_kernel(float* __restrict__ deg, int n) {
    int i = blockIdx.x * blockDim.x + threadIdx.x;
    if (i < n) deg[i] = rsqrtf(fmaxf(deg[i], 1.0f));
}

// ------------------------------ conversions --------------------------------

// fp32 features -> bf16, zero-fill padded rows (rows in [n, PAD))
__global__ void cvt_feat_kernel(const float* __restrict__ X, __bf16* __restrict__ A,
                                int n, int PAD, int F) {
    int i = blockIdx.x * blockDim.x + threadIdx.x;     // quad index
    int fq = F >> 2;
    if (i >= PAD * fq) return;
    int row = i / fq;
    int col = (i - row * fq) << 2;
    v4bf o;
    if (row < n) {
        float4 v = *(const float4*)&X[(size_t)row * F + col];
        o[0] = (__bf16)v.x; o[1] = (__bf16)v.y; o[2] = (__bf16)v.z; o[3] = (__bf16)v.w;
    } else {
        o[0] = o[1] = o[2] = o[3] = (__bf16)0.0f;
    }
    *(v4bf*)&A[(size_t)row * F + col] = o;
}

// W [K][N] fp32 -> Wt [N][K] bf16  (pre-transposed so GEMM B-loads are contiguous)
__global__ void cvt_w_kernel(const float* __restrict__ W, __bf16* __restrict__ Wt,
                             int K, int N) {
    int i = blockIdx.x * blockDim.x + threadIdx.x;
    if (i >= K * N) return;
    int nn = i / K, kk = i - nn * K;                   // consecutive i -> consecutive k
    Wt[(size_t)nn * K + kk] = (__bf16)W[(size_t)kk * N + nn];
}

// --------------------------------- GEMM ------------------------------------
// H[m][n] = out_norm[m] * sum_k A[m][k] * Wt[n][k]
// block: 128 threads = 4 waves; block tile M=128, N=64; wave tile 32x64.

#define TM 128
#define TN 64
#define TK 32
#define TKP 40   // padded K-stride in LDS (bank-conflict avoidance)

__global__ __launch_bounds__(128) void gemm_wmma_kernel(
    const __bf16* __restrict__ A,    // [PAD][F_in]  row-major bf16
    const __bf16* __restrict__ Wt,   // [F_out][F_in] row-major bf16 (W transposed)
    const float*  __restrict__ out_norm,
    float* __restrict__ H,           // [PAD][F_out] fp32
    int F_in, int F_out, int n_nodes)
{
    __shared__ __bf16 sA[2][TM][TKP];   // 2 x 10 KB
    __shared__ __bf16 sB[2][TN][TKP];   // 2 x  5 KB

    const int tid  = threadIdx.x;
    const int wave = tid >> 5;          // 0..3
    const int lane = tid & 31;
    const int half = lane >> 4;         // 0: lanes 0-15, 1: lanes 16-31
    const int l16  = lane & 15;
    const int mblk = blockIdx.x * TM;
    const int nblk = blockIdx.y * TN;

    v8f acc[2][4];
#pragma unroll
    for (int m = 0; m < 2; ++m)
#pragma unroll
        for (int nb = 0; nb < 4; ++nb) acc[m][nb] = (v8f)0.0f;

    // register staging for the next tile (A: 4 chunks, B: 2 chunks per thread)
    v8bf ga[4], gb[2];
    int aRow[4], aOff[4], bRow[2], bOff[2];
#pragma unroll
    for (int i = 0; i < 4; ++i) {
        int c = tid + i * 128;          // 0..511 -> 128 rows x 4 chunks
        aRow[i] = c >> 2;
        aOff[i] = (c & 3) * 8;
    }
#pragma unroll
    for (int i = 0; i < 2; ++i) {
        int c = tid + i * 128;          // 0..255 -> 64 rows x 4 chunks
        bRow[i] = c >> 2;
        bOff[i] = (c & 3) * 8;
    }

    // prologue: fetch tile k0=0 into registers
#pragma unroll
    for (int i = 0; i < 4; ++i)
        ga[i] = *(const v8bf*)&A[(size_t)(mblk + aRow[i]) * F_in + aOff[i]];
#pragma unroll
    for (int i = 0; i < 2; ++i)
        gb[i] = *(const v8bf*)&Wt[(size_t)(nblk + bRow[i]) * F_in + bOff[i]];

    int p = 0;
    for (int k0 = 0; k0 < F_in; k0 += TK, p ^= 1) {
        // commit staged registers into LDS buffer p
#pragma unroll
        for (int i = 0; i < 4; ++i) *(v8bf*)&sA[p][aRow[i]][aOff[i]] = ga[i];
#pragma unroll
        for (int i = 0; i < 2; ++i) *(v8bf*)&sB[p][bRow[i]][bOff[i]] = gb[i];
        __syncthreads();

        // prefetch next tile while the matrix units chew on buffer p
        int kn = k0 + TK;
        if (kn < F_in) {
#pragma unroll
            for (int i = 0; i < 4; ++i)
                ga[i] = *(const v8bf*)&A[(size_t)(mblk + aRow[i]) * F_in + kn + aOff[i]];
#pragma unroll
            for (int i = 0; i < 2; ++i)
                gb[i] = *(const v8bf*)&Wt[(size_t)(nblk + bRow[i]) * F_in + kn + bOff[i]];
        }

        // A fragments (two 16-row sub-tiles of the wave's 32 rows)
        // lane holds row M=l16; K = half*8+[0..7] and 16+half*8+[0..7]
        v16bf af[2];
#pragma unroll
        for (int m = 0; m < 2; ++m) {
            const __bf16* ar = &sA[p][wave * 32 + m * 16 + l16][0];
            af[m] = mk16(*(const v8bf*)&ar[half * 8], *(const v8bf*)&ar[16 + half * 8]);
        }
        // B fragments: lane holds col N=l16; K = half*8+[0..7] and 16+half*8+[0..7];
        // each fragment feeds two WMMAs (both A sub-tiles)
#pragma unroll
        for (int nb = 0; nb < 4; ++nb) {
            const __bf16* br = &sB[p][nb * 16 + l16][0];
            v16bf bf = mk16(*(const v8bf*)&br[half * 8], *(const v8bf*)&br[16 + half * 8]);
            acc[0][nb] = __builtin_amdgcn_wmma_f32_16x16x32_bf16(
                false, af[0], false, bf, (short)0, acc[0][nb], false, false);
            acc[1][nb] = __builtin_amdgcn_wmma_f32_16x16x32_bf16(
                false, af[1], false, bf, (short)0, acc[1][nb], false, false);
        }
        // double-buffered: the single barrier above is sufficient; next
        // iteration writes the other buffer.
    }

    // store: VGPR r of lane -> (M = half*8 + r, N = l16), scaled by out_norm[row]
#pragma unroll
    for (int m = 0; m < 2; ++m) {
        const int row0 = mblk + wave * 32 + m * 16 + half * 8;
#pragma unroll
        for (int nb = 0; nb < 4; ++nb) {
            int col = nblk + nb * 16 + l16;
#pragma unroll
            for (int r = 0; r < 8; ++r) {
                int row = row0 + r;
                if (row < n_nodes)
                    H[(size_t)row * F_out + col] = acc[m][nb][r] * out_norm[row];
            }
        }
    }
}

// --------------------------- edge scatter-add ------------------------------
// AGG[dst[e]][:] += H[src[e]][:]; consecutive threads per edge ->
// coalesced float4 gather, 4x global_atomic_add_f32 scatter (L2-resident).

__global__ void scatter_kernel(const int* __restrict__ src, const int* __restrict__ dst,
                               const float* __restrict__ H, float* __restrict__ AGG,
                               int E, int F) {
    int gid = blockIdx.x * blockDim.x + threadIdx.x;
    int fq = F >> 2;
    if (gid >= E * fq) return;
    int e = gid / fq;
    int c = (gid - e * fq) << 2;
    int s = src[e], d = dst[e];
    float4 v = *(const float4*)&H[(size_t)s * F + c];
    float* p = &AGG[(size_t)d * F + c];
    atomicAdd(p + 0, v.x);
    atomicAdd(p + 1, v.y);
    atomicAdd(p + 2, v.z);
    atomicAdd(p + 3, v.w);
}

// ------------------------------ epilogues ----------------------------------

// hidden layers: relu(AGG*in_norm + b) -> bf16 A for next layer (zero pad rows)
__global__ void post_relu_kernel(const float* __restrict__ AGG, const float* __restrict__ in_norm,
                                 const float* __restrict__ bias, __bf16* __restrict__ Aout,
                                 int n, int PAD, int F) {
    int i = blockIdx.x * blockDim.x + threadIdx.x;
    int fq = F >> 2;
    if (i >= PAD * fq) return;
    int row = i / fq;
    int col = (i - row * fq) << 2;
    v4bf o;
    if (row < n) {
        float4 a = *(const float4*)&AGG[(size_t)row * F + col];
        float4 b = *(const float4*)&bias[col];
        float  s = in_norm[row];
        o[0] = (__bf16)fmaxf(a.x * s + b.x, 0.0f);
        o[1] = (__bf16)fmaxf(a.y * s + b.y, 0.0f);
        o[2] = (__bf16)fmaxf(a.z * s + b.z, 0.0f);
        o[3] = (__bf16)fmaxf(a.w * s + b.w, 0.0f);
    } else {
        o[0] = o[1] = o[2] = o[3] = (__bf16)0.0f;
    }
    *(v4bf*)&Aout[(size_t)row * F + col] = o;
}

// final layer: AGG*in_norm + b -> fp32 output (no activation)
__global__ void post_final_kernel(const float* __restrict__ AGG, const float* __restrict__ in_norm,
                                  const float* __restrict__ bias, float* __restrict__ out,
                                  int n, int F) {
    int i = blockIdx.x * blockDim.x + threadIdx.x;
    int fq = F >> 2;
    if (i >= n * fq) return;
    int row = i / fq;
    int col = (i - row * fq) << 2;
    float4 a = *(const float4*)&AGG[(size_t)row * F + col];
    float4 b = *(const float4*)&bias[col];
    float  s = in_norm[row];
    float4 o;
    o.x = a.x * s + b.x;
    o.y = a.y * s + b.y;
    o.z = a.z * s + b.z;
    o.w = a.w * s + b.w;
    *(float4*)&out[(size_t)row * F + col] = o;
}

// ------------------------------- launcher ----------------------------------

extern "C" void kernel_launch(void* const* d_in, const int* in_sizes, int n_in,
                              void* d_out, int out_size, void* d_ws, size_t ws_size,
                              hipStream_t stream) {
    const float* features = (const float*)d_in[0];
    const int*   src      = (const int*)  d_in[1];
    const int*   dst      = (const int*)  d_in[2];
    const float* W0       = (const float*)d_in[3];
    const float* b0       = (const float*)d_in[4];
    const float* W1       = (const float*)d_in[5];
    const float* b1       = (const float*)d_in[6];
    const float* W2       = (const float*)d_in[7];
    const float* b2       = (const float*)d_in[8];

    const int F_IN = 512, F_HID = 512, F_OUT = 256;
    const int n   = in_sizes[0] / F_IN;          // 50000
    const int E   = in_sizes[1];                 // 800000
    const int PAD = ((n + TM - 1) / TM) * TM;    // 50048

    // workspace carve-out (~257 MB)
    char* w = (char*)d_ws;
    auto carve = [&](size_t bytes) -> char* {
        char* p = w; w += (bytes + 255) & ~(size_t)255; return p;
    };
    __bf16* Abuf = (__bf16*)carve((size_t)PAD * 512 * sizeof(__bf16));
    float*  H    = (float*) carve((size_t)PAD * 512 * sizeof(float));
    float*  AGG  = (float*) carve((size_t)PAD * 512 * sizeof(float));
    __bf16* Wt   = (__bf16*)carve((size_t)512 * 512 * sizeof(__bf16));
    float*  onrm = (float*) carve((size_t)PAD * sizeof(float));
    float*  inrm = (float*) carve((size_t)PAD * sizeof(float));
    (void)ws_size; (void)n_in; (void)out_size;

    const int TPB = 256;
    auto blocks = [&](long long work) -> int { return (int)((work + TPB - 1) / TPB); };

    // degrees -> norms
    hipMemsetAsync(onrm, 0, (size_t)PAD * sizeof(float), stream);
    hipMemsetAsync(inrm, 0, (size_t)PAD * sizeof(float), stream);
    degree_kernel<<<blocks(E), TPB, 0, stream>>>(src, dst, onrm, inrm, E);
    norm_kernel<<<blocks(PAD), TPB, 0, stream>>>(onrm, PAD);
    norm_kernel<<<blocks(PAD), TPB, 0, stream>>>(inrm, PAD);

    // features -> bf16 (padded)
    cvt_feat_kernel<<<blocks((long long)PAD * (F_IN / 4)), TPB, 0, stream>>>(
        features, Abuf, n, PAD, F_IN);

    struct Layer { const float* W; const float* b; int fin; int fout; };
    Layer L[3] = { {W0, b0, F_IN,  F_HID},
                   {W1, b1, F_HID, F_HID},
                   {W2, b2, F_HID, F_OUT} };

    for (int l = 0; l < 3; ++l) {
        const int fin = L[l].fin, fout = L[l].fout;
        cvt_w_kernel<<<blocks((long long)fin * fout), TPB, 0, stream>>>(L[l].W, Wt, fin, fout);

        dim3 grid(PAD / TM, fout / TN);
        gemm_wmma_kernel<<<grid, 128, 0, stream>>>(Abuf, Wt, onrm, H, fin, fout, n);

        hipMemsetAsync(AGG, 0, (size_t)PAD * fout * sizeof(float), stream);
        scatter_kernel<<<blocks((long long)E * (fout / 4)), TPB, 0, stream>>>(
            src, dst, H, AGG, E, fout);

        if (l < 2) {
            post_relu_kernel<<<blocks((long long)PAD * (fout / 4)), TPB, 0, stream>>>(
                AGG, inrm, L[l].b, Abuf, n, PAD, fout);
        } else {
            post_final_kernel<<<blocks((long long)n * (fout / 4)), TPB, 0, stream>>>(
                AGG, inrm, L[l].b, (float*)d_out, n, fout);
        }
    }
}
